// Predict2ndOrderTensor_47390669144217
// MI455X (gfx1250) — matrix operand order repaired and checked by reading, not compiled, not run on hardware
//
#include <hip/hip_runtime.h>
#include <stdint.h>

typedef __attribute__((ext_vector_type(16))) _Float16 v16h;
typedef __attribute__((ext_vector_type(2)))  __fp16   v2fp16;
typedef __attribute__((ext_vector_type(8)))  float    v8f;

union FragA { v16h h; uint32_t w[8]; uint4 q[2]; };

#define SH1 136   // halves stride, layer-1 input tile (128 cols + pad)
#define SH2 72    // halves stride, layer-2 input tile (64 cols + pad)
#define SF3 66    // float stride, layer-3 tile: conflict-free stores AND 32-lane dot reads
#define WS1 136   // halves stride of transposed W1 [64][128+pad]
#define WS2 72    // halves stride of transposed W2 [64][64+pad]

__device__ __forceinline__ float softplus_f(float x) {
  return fmaxf(x, 0.0f) + __logf(1.0f + __expf(-fabsf(x)));
}

// f32 -> packed f16x2 (v_cvt_pk_rtz_f16_f32)
__device__ __forceinline__ uint32_t pk2h(float a, float b) {
  union { v2fp16 h; uint32_t u; } p;
  p.h = __builtin_amdgcn_cvt_pkrtz(a, b);
  return p.u;
}

__global__ __launch_bounds__(256) void convert_msgs_kernel(
    const float* __restrict__ in, _Float16* __restrict__ out, int nQuads)
{
  int i = blockIdx.x * blockDim.x + threadIdx.x;   // 4 floats per thread
  if (i < nQuads) {
    float4 f = ((const float4*)in)[i];
    uint2 u;
    u.x = pk2h(f.x, f.y);
    u.y = pk2h(f.z, f.w);
    ((uint2*)out)[i] = u;
  }
}

template<bool USE_F16_MSGS>
__global__ __launch_bounds__(256) void triplet_mlp_pool_kernel(
    const float* __restrict__ messages, const _Float16* __restrict__ msgs16,
    const float* __restrict__ r,
    const float* __restrict__ W1, const float* __restrict__ b1,
    const float* __restrict__ W2, const float* __restrict__ b2,
    const float* __restrict__ W3, const float* __restrict__ b3,
    const int* __restrict__ idx_kj, const int* __restrict__ idx_ji,
    const int* __restrict__ idx_j,
    int T, int numTiles, int wavesTotal,
    float* __restrict__ accS, float* __restrict__ accC)
{
  __shared__ _Float16 sWt1[64 * WS1];          // W1^T as f16: [n][k], k<128
  __shared__ _Float16 sWt2[64 * WS2];          // W2^T as f16: [n][k], k<64
  __shared__ float sB1[64], sB2[64], sW3[64], sB3[1];
  __shared__ uint4 sStage[8 * 272];            // 8 waves x 4352B stage buffer

  const int tid = threadIdx.x;

  // ---- stage weights into LDS (f32 -> f16, transposed for B-fragments) ----
  for (int i = tid; i < 128 * 64; i += 256) {
    int k = i >> 6, n = i & 63;
    sWt1[n * WS1 + k] = (_Float16)W1[i];
  }
  for (int i = tid; i < 64 * 64; i += 256) {
    int k = i >> 6, n = i & 63;
    sWt2[n * WS2 + k] = (_Float16)W2[i];
  }
  if (tid < 64) { sB1[tid] = b1[tid]; sB2[tid] = b2[tid]; sW3[tid] = W3[tid]; }
  if (tid == 0) sB3[0] = b3[0];
  __syncthreads();

  const int lane = tid & 31;
  const int wave = tid >> 5;
  const int hi   = lane >> 4;   // half-wave selector per WMMA layouts
  const int ln   = lane & 15;
  _Float16* stage  = (_Float16*)(sStage + wave * 272);
  float*    stagef = (float*)stage;

  const int gwave = blockIdx.x * 8 + wave;

  for (int tile = gwave; tile < numTiles; tile += wavesTotal) {
    const int tbase = tile * 16;

    // ---- gather h = [m_kj | m_ji] (16 x 128) into f16 LDS tile ----
    {
      int row  = lane >> 1;                 // 0..15 : triplet row in tile
      int half = lane & 1;                  // 0: kj-half, 1: ji-half
      int gt  = tbase + row;
      int gtc = gt < T ? gt : T - 1;
      int src = half ? idx_ji[gtc] : idx_kj[gtc];
      if (USE_F16_MSGS) {
        const uint4* mp = (const uint4*)(msgs16 + (size_t)src * 64);
        uint4* dst = (uint4*)(stage + row * SH1 + half * 64);
        #pragma unroll
        for (int i = 0; i < 8; ++i) dst[i] = mp[i];
      } else {
        const float4* mp = (const float4*)(messages + (size_t)src * 64);
        _Float16* dst = stage + row * SH1 + half * 64;
        #pragma unroll
        for (int i = 0; i < 16; ++i) {
          float4 f = mp[i];
          uint2 u;
          u.x = pk2h(f.x, f.y);
          u.y = pk2h(f.z, f.w);
          *(uint2*)(dst + 4 * i) = u;
        }
      }
    }

    // ---- layer 1: [16x128] @ [128x64] via 16x v_wmma_f32_16x16x32_f16 ----
    FragA a1[4];
    #pragma unroll
    for (int f = 0; f < 4; ++f) {         // A frag: V0-3 K=base..+7, V4-7 K=base+16..+23
      const _Float16* base = stage + ln * SH1 + f * 32 + hi * 8;
      a1[f].q[0] = *(const uint4*)base;
      a1[f].q[1] = *(const uint4*)(base + 16);
    }
    v8f d1[4];
    #pragma unroll
    for (int nt = 0; nt < 4; ++nt) {
      float bv = sB1[nt * 16 + ln];
      v8f acc;
      #pragma unroll
      for (int v = 0; v < 8; ++v) acc[v] = bv;   // bias broadcast into C
      #pragma unroll
      for (int ks = 0; ks < 4; ++ks) {           // B frag: V0-3 K=base..+7, V4-7 K=base+16..+23
        FragA b;
        const _Float16* base = sWt1 + (nt * 16 + ln) * WS1 + ks * 32 + hi * 8;
        b.q[0] = *(const uint4*)base;
        b.q[1] = *(const uint4*)(base + 16);
        acc = __builtin_amdgcn_wmma_f32_16x16x32_f16(false, a1[ks].h, false, b.h,
                                                     (short)0, acc, false, false);
      }
      d1[nt] = acc;
    }
    // softplus -> restage h2 as f16 [16 x 64] (D layout: M = v + 8*hi, N = nt*16+ln)
    #pragma unroll
    for (int nt = 0; nt < 4; ++nt) {
      #pragma unroll
      for (int v = 0; v < 8; ++v) {
        int m = v + hi * 8;
        stage[m * SH2 + nt * 16 + ln] = (_Float16)softplus_f(d1[nt][v]);
      }
    }

    // ---- layer 2: [16x64] @ [64x64] via 8 WMMAs ----
    FragA a2[2];
    #pragma unroll
    for (int f = 0; f < 2; ++f) {
      const _Float16* base = stage + ln * SH2 + f * 32 + hi * 8;
      a2[f].q[0] = *(const uint4*)base;
      a2[f].q[1] = *(const uint4*)(base + 16);
    }
    #pragma unroll
    for (int nt = 0; nt < 4; ++nt) {
      float bv = sB2[nt * 16 + ln];
      v8f acc;
      #pragma unroll
      for (int v = 0; v < 8; ++v) acc[v] = bv;
      #pragma unroll
      for (int ks = 0; ks < 2; ++ks) {
        FragA b;
        const _Float16* base = sWt2 + (nt * 16 + ln) * WS2 + ks * 32 + hi * 8;
        b.q[0] = *(const uint4*)base;
        b.q[1] = *(const uint4*)(base + 16);
        acc = __builtin_amdgcn_wmma_f32_16x16x32_f16(false, a2[ks].h, false, b.h,
                                                     (short)0, acc, false, false);
      }
      // softplus -> stage h3 as f32 with stride SF3 (conflict-free)
      #pragma unroll
      for (int v = 0; v < 8; ++v) {
        int m = v + hi * 8;
        stagef[m * SF3 + nt * 16 + ln] = softplus_f(acc[v]);
      }
    }

    // ---- layer 3 (64->1): 32-lane split dot; lane (ln,hi) does k in [32*hi, 32*hi+32) ----
    float b3v = sB3[0];
    float mv = hi ? 0.0f : b3v;
    {
      const float* h3 = stagef + ln * SF3 + hi * 32;
      const float* w3 = sW3 + hi * 32;
      #pragma unroll
      for (int k = 0; k < 32; ++k) mv += h3[k] * w3[k];
      mv += __shfl_xor(mv, 16, 32);   // both half-waves now hold full dot for triplet ln
    }

    // ---- dyad + edge-grouped segment-sum atomics (lanes 0..15) ----
    bool fullTile = (tbase + 16) <= T;
    if (lane < 16) {
      int gt = tbase + lane;
      if (gt < T) {
        int j = idx_j[gt];
        float2 rk = *(const float2*)(r + 2 * (size_t)idx_kj[gt]);
        float2 rj = *(const float2*)(r + 2 * (size_t)idx_ji[gt]);
        float t0 = mv * rj.x * rk.x;
        float t1 = mv * rj.x * rk.y;
        float t2 = mv * rj.y * rk.x;
        float t3 = mv * rj.y * rk.y;
        if (fullTile) {
          // 8 triplets of one edge share idx_j -> reduce groups of 8 in-wave
          #pragma unroll
          for (int msk = 1; msk <= 4; msk <<= 1) {
            t0 += __shfl_xor(t0, msk, 32);
            t1 += __shfl_xor(t1, msk, 32);
            t2 += __shfl_xor(t2, msk, 32);
            t3 += __shfl_xor(t3, msk, 32);
          }
          if ((lane & 7) == 0) {
            atomicAdd(&accS[(size_t)j * 4 + 0], t0);
            atomicAdd(&accS[(size_t)j * 4 + 1], t1);
            atomicAdd(&accS[(size_t)j * 4 + 2], t2);
            atomicAdd(&accS[(size_t)j * 4 + 3], t3);
            atomicAdd(&accC[j], 8.0f);
          }
        } else {
          atomicAdd(&accS[(size_t)j * 4 + 0], t0);
          atomicAdd(&accS[(size_t)j * 4 + 1], t1);
          atomicAdd(&accS[(size_t)j * 4 + 2], t2);
          atomicAdd(&accS[(size_t)j * 4 + 3], t3);
          atomicAdd(&accC[j], 1.0f);
        }
      }
    }
  }
}

__global__ __launch_bounds__(256) void finalize_mean_kernel(
    const float* __restrict__ accS, const float* __restrict__ accC,
    float* __restrict__ out, int n4)
{
  int i = blockIdx.x * blockDim.x + threadIdx.x;
  if (i < n4) {
    float c = accC[i >> 2];
    out[i] = accS[i] / fmaxf(c, 1.0f);
  }
}

extern "C" void kernel_launch(void* const* d_in, const int* in_sizes, int n_in,
                              void* d_out, int out_size, void* d_ws, size_t ws_size,
                              hipStream_t stream) {
  (void)n_in;
  const float* messages = (const float*)d_in[0];
  const float* r  = (const float*)d_in[1];
  const float* W1 = (const float*)d_in[2];
  const float* b1 = (const float*)d_in[3];
  const float* W2 = (const float*)d_in[4];
  const float* b2 = (const float*)d_in[5];
  const float* W3 = (const float*)d_in[6];
  const float* b3 = (const float*)d_in[7];
  const int* idx_kj = (const int*)d_in[8];
  const int* idx_ji = (const int*)d_in[9];
  const int* idx_j  = (const int*)d_in[10];

  int T = in_sizes[8];            // number of triplets
  int N = out_size / 4;           // nodes (out is [N,2,2])
  int E = in_sizes[0] / 64;       // edges (messages rows)

  float* accS = (float*)d_ws;             // [N*4] sums
  float* accC = accS + (size_t)N * 4;     // [N]   counts
  (void)hipMemsetAsync(d_ws, 0, sizeof(float) * (size_t)N * 5, stream);

  // f16 message cache in workspace (L2-resident: ~20.5 MB for E=160K)
  size_t msgOff = ((size_t)N * 5 * sizeof(float) + 255) & ~(size_t)255;
  size_t need   = msgOff + (size_t)E * 64 * sizeof(_Float16);
  bool useF16   = (ws_size >= need) && (E > 0);
  _Float16* msgs16 = (_Float16*)((char*)d_ws + msgOff);

  int numTiles = (T + 15) / 16;
  int blocks = (numTiles + 7) / 8;
  if (blocks > 320) blocks = 320;
  if (blocks < 1)   blocks = 1;
  int wavesTotal = blocks * 8;

  if (useF16) {
    int nQuads = E * 16;  // E*64 floats / 4
    convert_msgs_kernel<<<(nQuads + 255) / 256, 256, 0, stream>>>(messages, msgs16, nQuads);
    triplet_mlp_pool_kernel<true><<<blocks, 256, 0, stream>>>(
        messages, msgs16, r, W1, b1, W2, b2, W3, b3, idx_kj, idx_ji, idx_j,
        T, numTiles, wavesTotal, accS, accC);
  } else {
    triplet_mlp_pool_kernel<false><<<blocks, 256, 0, stream>>>(
        messages, msgs16, r, W1, b1, W2, b2, W3, b3, idx_kj, idx_ji, idx_j,
        T, numTiles, wavesTotal, accS, accC);
  }

  int n4 = N * 4;
  finalize_mean_kernel<<<(n4 + 255) / 256, 256, 0, stream>>>(accS, accC,
                                                             (float*)d_out, n4);
}
